// ExpertPredictionHeads_89094801588572
// MI455X (gfx1250) — matrix operand order repaired and checked by reading, not compiled, not run on hardware
//
#include <hip/hip_runtime.h>

// ---------------- problem constants ----------------
#define EN      262144
#define DN      512
#define DE      64
#define DIN     1088          // 2*DN + DE
#define DH      256
#define KEXP    3
#define MROWS   32            // edges per block (2 WMMA M-tiles)
#define TILES32 (EN / MROWS)  // 8192 max tiles per expert
#define KS1     (DIN / 32)    // 34 K-steps, layer 1
#define KS2     (DH / 32)     // 8  K-steps, layer 2
#define NT      (DH / 16)     // 16 N-tiles of 16 cols
#define W1DW    (KEXP * KS1 * NT * 256)   // dwords of repacked W1 fragments
#define W2DW    (KEXP * KS2 * NT * 256)   // dwords of repacked W2 fragments

typedef __attribute__((ext_vector_type(16))) __bf16 v16bf;
typedef __attribute__((ext_vector_type(8)))  float  v8f;

union Frag { uint4 q[2]; v16bf v; };

__device__ __forceinline__ unsigned short f2bf(float x) {
    union { float f; unsigned u; } c; c.f = x;
    unsigned u = c.u;
    unsigned r = u + 0x7FFFu + ((u >> 16) & 1u);   // round-to-nearest-even
    return (unsigned short)(r >> 16);
}

__device__ __forceinline__ float gelu_exact(float x) {
    return 0.5f * x * (1.0f + erff(x * 0.7071067811865475f));
}

// CDNA5 async copy: per-lane 16B global -> LDS, tracked by ASYNCcnt.
__device__ __forceinline__ void async_copy16(unsigned lds_off, const void* gsrc) {
    unsigned long long ga = (unsigned long long)(uintptr_t)gsrc;
    asm volatile("global_load_async_to_lds_b128 %0, %1, off"
                 :: "v"(lds_off), "v"(ga) : "memory");
}
__device__ __forceinline__ void wait_async0() {
    asm volatile("s_wait_asynccnt 0x0" ::: "memory");
}

// ---------------- bucketing kernels ----------------
// ws header ints: [0..3]=counts, [4..7]=offsets, [8..11]=cursor
__global__ void init_kernel(int* hdr) {
    if (threadIdx.x < 16) hdr[threadIdx.x] = 0;
}

__global__ void hist_kernel(const int* __restrict__ etype, int* __restrict__ hdr) {
    int e = blockIdx.x * blockDim.x + threadIdx.x;
    if (e < EN) atomicAdd(&hdr[etype[e]], 1);
}

__global__ void scan_kernel(int* hdr) {
    if (threadIdx.x == 0) {
        int o = 0;
        for (int k = 0; k < KEXP; k++) { hdr[4 + k] = o; hdr[8 + k] = o; o += hdr[k]; }
    }
}

__global__ void scatter_kernel(const int* __restrict__ etype, int* __restrict__ hdr,
                               int* __restrict__ perm) {
    int e = blockIdx.x * blockDim.x + threadIdx.x;
    if (e < EN) {
        int k = etype[e];
        int p = atomicAdd(&hdr[8 + k], 1);
        perm[p] = e;
    }
}

// ---------------- weight repack: f32 -> bf16 WMMA B-fragments ----------------
// Layout: [k][ks][nt][256 dwords]  -> each K-step panel is one contiguous 16KB
// block (NT * 1KB), ready for async staging. Within a 32x16 (KxN) block:
// lane L owns 8 dwords at [lane*8..+7]; n = L%16, K half = (L/16)*16,
// dword j packs K = half+2j (lo 16b), half+2j+1 (hi 16b).
__global__ void repack_kernel(const float* __restrict__ W1, const float* __restrict__ W2,
                              unsigned* __restrict__ w1f, unsigned* __restrict__ w2f) {
    int id = blockIdx.x * blockDim.x + threadIdx.x;
    if (id < W1DW) {
        int j = id & 7, lane = (id >> 3) & 31, rest = id >> 8;
        int nt = rest % NT; rest /= NT;
        int ks = rest % KS1; int k = rest / KS1;
        int n  = nt * 16 + (lane & 15);
        int kk = ks * 32 + ((lane >> 4) << 4) + 2 * j;
        const float* Wk = W1 + (size_t)k * DIN * DH;
        unsigned lo = f2bf(Wk[(size_t)kk * DH + n]);
        unsigned hi = f2bf(Wk[(size_t)(kk + 1) * DH + n]);
        w1f[id] = lo | (hi << 16);
    } else if (id < W1DW + W2DW) {
        int id2 = id - W1DW;
        int j = id2 & 7, lane = (id2 >> 3) & 31, rest = id2 >> 8;
        int nt = rest % NT; rest /= NT;
        int ks = rest % KS2; int k = rest / KS2;
        int n  = nt * 16 + (lane & 15);
        int kk = ks * 32 + ((lane >> 4) << 4) + 2 * j;
        const float* Wk = W2 + (size_t)k * DH * DH;
        unsigned lo = f2bf(Wk[(size_t)kk * DH + n]);
        unsigned hi = f2bf(Wk[(size_t)(kk + 1) * DH + n]);
        w2f[id2] = lo | (hi << 16);
    }
}

// ---------------- main fused MoE-MLP kernel ----------------
// Block = 256 threads (8 waves) = 32 edges of one expert (2 M-tiles).
// Wave w: mtile = w>>2 (rows mtile*16..+15), colgroup = w&3 (4 N-tiles = 64 cols).
// B fragments double-buffered in LDS via global_load_async_to_lds_b128.
__global__ __launch_bounds__(256) void edge_moe_kernel(
    const float* __restrict__ node, const float* __restrict__ estate,
    const int*   __restrict__ eindex, const float* __restrict__ baseline,
    const float* __restrict__ b1, const float* __restrict__ b2,
    const float* __restrict__ W3, const float* __restrict__ b3,
    const int*   __restrict__ hdr, const int* __restrict__ perm,
    const unsigned* __restrict__ w1f, const unsigned* __restrict__ w2f,
    float* __restrict__ out)
{
    const int tid  = threadIdx.x;
    const int lane = tid & 31;
    const int wave = tid >> 5;
    const int k    = blockIdx.x / TILES32;
    const int t    = blockIdx.x % TILES32;

    const int cnt = hdr[k];
    if (t * MROWS >= cnt) return;                 // uniform exit: EXEC stays all-1s
    const int base   = hdr[4 + k] + t * MROWS;
    const int nvalid = (cnt - t * MROWS < MROWS) ? (cnt - t * MROWS) : MROWS;

    // U (bf16 bits) for 32 rows; H is aliased over it after layer 1.
    __shared__ alignas(16) unsigned short Ubuf[MROWS][DIN + 8];     // 70,144 B
    __shared__ alignas(16) unsigned char  Bstage[2][NT * 1024];     // 32,768 B
    __shared__ float dpart[4][MROWS];
    __shared__ int   eids[MROWS];
    unsigned short (*Hlds)[DH + 8] = (unsigned short (*)[DH + 8])&Ubuf[0][0];

    if (tid < MROWS) {
        int mm = (tid < nvalid) ? tid : (nvalid - 1);
        eids[tid] = perm[base + mm];
    }
    __syncthreads();

    // ---- gather u rows into LDS as bf16 ----
    for (int m = wave; m < MROWS; m += 8) {
        const int e = eids[m];
        const int s = eindex[e];
        const int d = eindex[EN + e];
        const float* ps = node + (size_t)s * DN;
        const float* pd = node + (size_t)d * DN;
        for (int c = lane * 4; c < DN; c += 128) {
            float4 v = *(const float4*)(ps + c);
            Ubuf[m][c + 0] = f2bf(v.x); Ubuf[m][c + 1] = f2bf(v.y);
            Ubuf[m][c + 2] = f2bf(v.z); Ubuf[m][c + 3] = f2bf(v.w);
            float4 w = *(const float4*)(pd + c);
            Ubuf[m][DN + c + 0] = f2bf(w.x); Ubuf[m][DN + c + 1] = f2bf(w.y);
            Ubuf[m][DN + c + 2] = f2bf(w.z); Ubuf[m][DN + c + 3] = f2bf(w.w);
        }
        float2 ev = *(const float2*)(estate + (size_t)e * DE + lane * 2);
        Ubuf[m][2 * DN + lane * 2 + 0] = f2bf(ev.x);
        Ubuf[m][2 * DN + lane * 2 + 1] = f2bf(ev.y);
    }

    const int row   = lane & 15;                  // A row within mtile / C col idx
    const int khalf = (lane >> 4) << 4;           // 0 or 16: A K-offset per lane half
    const int mrow0 = (lane < 16) ? 0 : 8;        // C/D row base per lane half
    const int mt    = wave >> 2;                  // M-tile of this wave
    const int cg    = wave & 3;                   // column group (4 N-tiles)
    const int arow  = mt * 16 + row;              // global A row in Ubuf/Hlds

    const unsigned bs0 = (unsigned)(uintptr_t)&Bstage[0][0];
    const unsigned bs1 = (unsigned)(uintptr_t)&Bstage[1][0];

    // stage one 16KB K-step panel (NT tiles) : 256 threads x 4 x 16B lanes
    const unsigned* Wb1 = w1f + (size_t)k * KS1 * NT * 256;
    {
        const char* src = (const char*)Wb1;
        #pragma unroll
        for (int i = 0; i < 4; i++) {
            int c = tid + i * 256;
            async_copy16(bs0 + (unsigned)c * 16u, src + (size_t)c * 16);
        }
    }
    wait_async0();
    __syncthreads();    // U gather + first B panel visible to all waves

    // ---- layer 1: C1[32x256] = U[32x1088] x W1[k] ----
    v8f acc[4];
    #pragma unroll
    for (int tt = 0; tt < 4; tt++) {
        #pragma unroll
        for (int i = 0; i < 8; i++) acc[tt][i] = 0.0f;
    }
    for (int ks = 0; ks < KS1; ks++) {
        const int buf = ks & 1;
        if (ks + 1 < KS1) {   // prefetch next panel into the other buffer
            const char* src = (const char*)(Wb1 + (size_t)(ks + 1) * NT * 256);
            unsigned dst = buf ? bs0 : bs1;
            #pragma unroll
            for (int i = 0; i < 4; i++) {
                int c = tid + i * 256;
                async_copy16(dst + (unsigned)c * 16u, src + (size_t)c * 16);
            }
        }
        Frag a;
        const uint4* ap = (const uint4*)&Ubuf[arow][ks * 32 + khalf];
        a.q[0] = ap[0];            // K base..base+7
        a.q[1] = ap[1];            // K base+8..base+15
        #pragma unroll
        for (int tt = 0; tt < 4; tt++) {
            const uint4* bp = (const uint4*)&Bstage[buf][(cg * 4 + tt) * 1024 + lane * 32];
            Frag b; b.q[0] = bp[0]; b.q[1] = bp[1];
            acc[tt] = __builtin_amdgcn_wmma_f32_16x16x32_bf16(
                false, a.v, false, b.v, (short)0, acc[tt], false, false);
        }
        if (ks + 1 < KS1) { wait_async0(); __syncthreads(); }
    }
    __syncthreads();   // all A reads of Ubuf done before H overlays it

    // bias + gelu, write H (bf16) into aliased LDS region
    #pragma unroll
    for (int tt = 0; tt < 4; tt++) {
        const int col  = (cg * 4 + tt) * 16 + row;
        const float bb = b1[k * DH + col];
        #pragma unroll
        for (int r = 0; r < 8; r++) {
            float x = gelu_exact(acc[tt][r] + bb);
            Hlds[mt * 16 + r + mrow0][col] = f2bf(x);
        }
    }
    __syncthreads();

    // ---- layer 2: C2[32x256] = H[32x256] x W2[k] ----
    v8f acc2[4];
    #pragma unroll
    for (int tt = 0; tt < 4; tt++) {
        #pragma unroll
        for (int i = 0; i < 8; i++) acc2[tt][i] = 0.0f;
    }
    const unsigned* Wb2 = w2f + (size_t)k * KS2 * NT * 256;
    {
        const char* src = (const char*)Wb2;
        #pragma unroll
        for (int i = 0; i < 4; i++) {
            int c = tid + i * 256;
            async_copy16(bs0 + (unsigned)c * 16u, src + (size_t)c * 16);
        }
    }
    wait_async0();
    __syncthreads();
    #pragma unroll
    for (int ks = 0; ks < KS2; ks++) {
        const int buf = ks & 1;
        if (ks + 1 < KS2) {
            const char* src = (const char*)(Wb2 + (size_t)(ks + 1) * NT * 256);
            unsigned dst = buf ? bs0 : bs1;
            #pragma unroll
            for (int i = 0; i < 4; i++) {
                int c = tid + i * 256;
                async_copy16(dst + (unsigned)c * 16u, src + (size_t)c * 16);
            }
        }
        Frag a;
        const uint4* ap = (const uint4*)&Hlds[arow][ks * 32 + khalf];
        a.q[0] = ap[0];
        a.q[1] = ap[1];
        #pragma unroll
        for (int tt = 0; tt < 4; tt++) {
            const uint4* bp = (const uint4*)&Bstage[buf][(cg * 4 + tt) * 1024 + lane * 32];
            Frag b; b.q[0] = bp[0]; b.q[1] = bp[1];
            acc2[tt] = __builtin_amdgcn_wmma_f32_16x16x32_bf16(
                false, a.v, false, b.v, (short)0, acc2[tt], false, false);
        }
        if (ks + 1 < KS2) { wait_async0(); __syncthreads(); }
    }

    // ---- layer 3: bias + gelu + dot with W3[k], lane-half butterfly reduce ----
    float p[8];
    #pragma unroll
    for (int r = 0; r < 8; r++) p[r] = 0.0f;
    #pragma unroll
    for (int tt = 0; tt < 4; tt++) {
        const int col  = (cg * 4 + tt) * 16 + row;
        const float bb = b2[k * DH + col];
        const float wv = W3[k * DH + col];
        #pragma unroll
        for (int r = 0; r < 8; r++) {
            float x = gelu_exact(acc2[tt][r] + bb);
            p[r] += x * wv;
        }
    }
    #pragma unroll
    for (int off = 8; off > 0; off >>= 1) {
        #pragma unroll
        for (int r = 0; r < 8; r++) p[r] += __shfl_xor(p[r], off, 16);
    }
    if (row == 0) {
        #pragma unroll
        for (int r = 0; r < 8; r++) dpart[cg][mt * 16 + r + mrow0] = p[r];
    }
    __syncthreads();

    if (tid < nvalid) {
        float dsum = dpart[0][tid] + dpart[1][tid] + dpart[2][tid] + dpart[3][tid] + b3[k];
        int e = eids[tid];
        out[e] = dsum;                                  // delta
        out[EN + e] = tanhf(baseline[e] + dsum);        // rho
    }
}

// ---------------- launch ----------------
extern "C" void kernel_launch(void* const* d_in, const int* in_sizes, int n_in,
                              void* d_out, int out_size, void* d_ws, size_t ws_size,
                              hipStream_t stream)
{
    const float* node     = (const float*)d_in[0];
    const float* estate   = (const float*)d_in[1];
    const int*   eindex   = (const int*)  d_in[2];
    const int*   etype    = (const int*)  d_in[3];
    const float* baseline = (const float*)d_in[4];
    const float* W1       = (const float*)d_in[5];
    const float* b1       = (const float*)d_in[6];
    const float* W2       = (const float*)d_in[7];
    const float* b2       = (const float*)d_in[8];
    const float* W3       = (const float*)d_in[9];
    const float* b3       = (const float*)d_in[10];
    float* out = (float*)d_out;

    char* ws = (char*)d_ws;
    int*      hdr  = (int*)ws;                         // 64 B header
    int*      perm = (int*)(ws + 64);                  // EN ints
    unsigned* w1f  = (unsigned*)(ws + 64 + (size_t)EN * 4);
    unsigned* w2f  = w1f + W1DW;
    // total ws use: 64 + 1 MiB + ~1.6 MiB + ~0.38 MiB  (~3.1 MB)

    init_kernel<<<1, 64, 0, stream>>>(hdr);
    hist_kernel<<<(EN + 255) / 256, 256, 0, stream>>>(etype, hdr);
    scan_kernel<<<1, 32, 0, stream>>>(hdr);
    scatter_kernel<<<(EN + 255) / 256, 256, 0, stream>>>(etype, hdr, perm);
    repack_kernel<<<(W1DW + W2DW + 255) / 256, 256, 0, stream>>>(W1, W2, w1f, w2f);
    edge_moe_kernel<<<KEXP * TILES32, 256, 0, stream>>>(
        node, estate, eindex, baseline, b1, b2, W3, b3, hdr, perm, w1f, w2f, out);
}
